// MultiHeadAttention_16733192585787
// MI455X (gfx1250) — compile-verified, matrix operand order checked by reading, not run on hardware
//
#include <hip/hip_runtime.h>
#include <hip/hip_bf16.h>
#include <stdint.h>

// ---------------------------------------------------------------------------
// MHA for MI455X (gfx1250, wave32, WMMA). bf16 inputs, f32 accumulate.
// B=2, S=2048, HID=1024, NH=16, HD=64
// ---------------------------------------------------------------------------

#define B_   2
#define S_   2048
#define HID_ 1024
#define NH_  16
#define HD_  64
#define M_   (B_ * S_)      // 4096 rows for the projection GEMMs
#define K_   HID_           // 1024

typedef __bf16 bf16;
typedef __attribute__((ext_vector_type(16))) __bf16 v16bf;
typedef __attribute__((ext_vector_type(4)))  __bf16 v4bf;
typedef __attribute__((ext_vector_type(8)))  float  v8f;

__device__ __forceinline__ v8f wmma_bf16(v16bf a, v16bf b, v8f c) {
  // D = A(16x32 bf16) x B(32x16 bf16) + C(16x16 f32)
  return __builtin_amdgcn_wmma_f32_16x16x32_bf16(
      /*neg_a=*/false, a, /*neg_b=*/false, b,
      /*c_mod=*/(short)0, c, /*reuse_a=*/false, /*reuse_b=*/false);
}

// Uniform fragment loader for a 16x32 bf16 tile of a row-major matrix.
// A-layout (M x K): lane l<16 -> row l, K {0..7, 16..23}; lane>=16 -> row l-16,
// K {8..15, 24..31}.  The B-layout (K x N) of a matrix equals the A-layout of
// its transpose, so B fragments are loaded from transposed (N x K) storage
// with this same function.  Two global_load_b128 per fragment.
__device__ __forceinline__ v16bf load_frag(const bf16* __restrict__ base, int ld) {
  const int lane = threadIdx.x & 31;
  const int row  = lane & 15;
  const int half = lane >> 4;               // 0 or 1
  const bf16* p = base + row * ld + half * 8;
  union { v16bf v; uint4 u[2]; } r;
  r.u[0] = *reinterpret_cast<const uint4*>(p);       // cols +0  (8 bf16)
  r.u[1] = *reinterpret_cast<const uint4*>(p + 16);  // cols +16 (8 bf16)
  return r.v;
}

// ---------------------------------------------------------------------------
// 1) f32 -> bf16 elementwise convert (vectorized x4)
// ---------------------------------------------------------------------------
__global__ void cvt_f32_bf16(const float* __restrict__ in,
                             bf16* __restrict__ out, int n) {
  int i = (blockIdx.x * blockDim.x + threadIdx.x) * 4;
  if (i + 3 < n) {
    float4 v = *reinterpret_cast<const float4*>(in + i);
    v4bf o;
    o.x = (bf16)v.x; o.y = (bf16)v.y; o.z = (bf16)v.z; o.w = (bf16)v.w;
    *reinterpret_cast<v4bf*>(out + i) = o;
  }
}

// ---------------------------------------------------------------------------
// 2) W[K x N] f32  ->  Wt[N x K] bf16  (32x32 LDS tile transpose)
// ---------------------------------------------------------------------------
__global__ void transpose_cvt(const float* __restrict__ W,
                              bf16* __restrict__ Wt) {
  __shared__ float tile[32][33];
  int x = blockIdx.x * 32 + threadIdx.x;  // n
  int y = blockIdx.y * 32 + threadIdx.y;  // k
  tile[threadIdx.y][threadIdx.x] = W[(size_t)y * HID_ + x];
  __syncthreads();
  int xo = blockIdx.y * 32 + threadIdx.x; // k
  int yo = blockIdx.x * 32 + threadIdx.y; // n
  Wt[(size_t)yo * HID_ + xo] = (bf16)tile[threadIdx.x][threadIdx.y];
}

// ---------------------------------------------------------------------------
// 3) GEMM: C[M x N] = A[M x K] * W[K x N] + bias, via Wt[N x K].
//    4 waves / WG; each wave computes a 32(M) x 64(N) tile = 8 WMMA C tiles.
//    MODE is a compile-time parameter -> straight-line, branch-free epilogue:
//      0: store bf16 head-split     out[((b*NH+h)*S + s)*HD + d]   (Q, K)
//      1: store bf16 head-split, V transposed  out[((b*NH+h)*HD + d)*S + s]
//      2: store f32 row-major       out[m*HID + n]                 (final)
// ---------------------------------------------------------------------------
template <int MODE>
__global__ void gemm_wmma(const bf16* __restrict__ A,
                          const bf16* __restrict__ Wt,
                          const float* __restrict__ bias,
                          void* __restrict__ out) {
  const int wave = threadIdx.x >> 5;
  const int m0 = (blockIdx.x * 4 + wave) * 32;
  const int n0 = blockIdx.y * 64;

  v8f c[2][4] = {};
  for (int k = 0; k < K_; k += 32) {
    v16bf a0 = load_frag(A + (size_t)m0 * K_ + k, K_);
    v16bf a1 = load_frag(A + (size_t)(m0 + 16) * K_ + k, K_);
    v16bf b[4];
#pragma unroll
    for (int t = 0; t < 4; ++t)
      b[t] = load_frag(Wt + (size_t)(n0 + t * 16) * K_ + k, K_);
#pragma unroll
    for (int t = 0; t < 4; ++t) {
      c[0][t] = wmma_bf16(a0, b[t], c[0][t]);
      c[1][t] = wmma_bf16(a1, b[t], c[1][t]);
    }
  }

  // Epilogue.  C layout: lane l, elem r -> (m_local = r + 8*(l>=16), n_local = l&15)
  const int lane   = threadIdx.x & 31;
  const int rowadd = (lane & 16) ? 8 : 0;
  const int nloc   = lane & 15;
#pragma unroll
  for (int i = 0; i < 2; ++i) {
#pragma unroll
    for (int t = 0; t < 4; ++t) {
      const int n = n0 + t * 16 + nloc;
      const float bv = bias[n];
      const int mb = m0 + i * 16 + rowadd;          // base row for this tile
      if (MODE == 2) {
        float* o = reinterpret_cast<float*>(out) + (size_t)mb * HID_ + n;
#pragma unroll
        for (int r = 0; r < 8; ++r) o[(size_t)r * HID_] = c[i][t][r] + bv;
      } else {
        const int bb = mb >> 11, s = mb & (S_ - 1);   // rows r stay in-batch
        const int h = n >> 6, d = n & (HD_ - 1);
        bf16* o;
        size_t stride;
        if (MODE == 0) {  // [B, NH, S, HD], consecutive r -> +HD along s
          o = reinterpret_cast<bf16*>(out) +
              ((((size_t)bb * NH_ + h) * S_ + s) * HD_ + d);
          stride = HD_;
        } else {          // [B, NH, HD, S], consecutive r -> +1 along s
          o = reinterpret_cast<bf16*>(out) +
              ((((size_t)bb * NH_ + h) * HD_ + d) * S_ + s);
          stride = 1;
        }
#pragma unroll
        for (int r = 0; r < 8; ++r) o[(size_t)r * stride] = (bf16)(c[i][t][r] + bv);
      }
    }
  }
}

// ---------------------------------------------------------------------------
// 4) Flash attention, one wave per 16 query rows of one head.
//    Score tiles computed TRANSPOSED (S' = K*Q^T: M=keys, N=queries) so that
//    the two stacked 16x16 C tiles of a 32-key block are, register-for-
//    register, the A-layout of the 16x32 bf16 P^T fragment for ctx = P'^T * V.
//    Softmax stats per query are lane-local (query = lane%16, replicated in
//    both half-waves); only shfl_xor(16) reductions + 8 broadcasts needed.
// ---------------------------------------------------------------------------
__global__ void attn_wmma(const bf16* __restrict__ Qb,
                          const bf16* __restrict__ Kb,
                          const bf16* __restrict__ Vtb,
                          bf16* __restrict__ ctx) {
  const int head = blockIdx.y;              // b*NH + h
  const int bb = head >> 4, h = head & 15;
  const int q0 = blockIdx.x * 16;
  const bf16* Q  = Qb  + (size_t)head * S_ * HD_;
  const bf16* Kp = Kb  + (size_t)head * S_ * HD_;
  const bf16* Vt = Vtb + (size_t)head * HD_ * S_;  // [HD x S] per head
  const int lane = threadIdx.x & 31;

  // Q fragments (B-side of S'), loop invariant: d 0..31 and 32..63.
  const v16bf bq0 = load_frag(Q + (size_t)q0 * HD_, HD_);
  const v16bf bq1 = load_frag(Q + (size_t)q0 * HD_ + 32, HD_);

  float m_run = -1e30f, l_run = 0.0f;
  v8f acc[4] = {};                          // ctx tiles, d-tiles 0..3
  const float scale = 0.125f;               // 1/sqrt(64)
  const int srcbase = (lane & 16) ? 8 : 0;

  for (int n0 = 0; n0 < S_; n0 += 32) {
    if (n0 + 32 < S_)
      __builtin_prefetch(Kp + (size_t)(n0 + 32) * HD_, 0, 1);

    // S' tiles: t0 = keys n0..n0+15, t1 = keys n0+16..n0+31 (M=key, N=query)
    v16bf ak0a = load_frag(Kp + (size_t)n0 * HD_, HD_);
    v16bf ak0b = load_frag(Kp + (size_t)n0 * HD_ + 32, HD_);
    v16bf ak1a = load_frag(Kp + (size_t)(n0 + 16) * HD_, HD_);
    v16bf ak1b = load_frag(Kp + (size_t)(n0 + 16) * HD_ + 32, HD_);
    v8f t0 = {}; t0 = wmma_bf16(ak0a, bq0, t0); t0 = wmma_bf16(ak0b, bq1, t0);
    v8f t1 = {}; t1 = wmma_bf16(ak1a, bq0, t1); t1 = wmma_bf16(ak1b, bq1, t1);

    // Online softmax (per query = lane%16; lanes l and l^16 are replicas).
    float mx = -1e30f;
#pragma unroll
    for (int r = 0; r < 8; ++r) {
      t0[r] *= scale; t1[r] *= scale;
      mx = fmaxf(mx, fmaxf(t0[r], t1[r]));
    }
    mx = fmaxf(mx, __shfl_xor(mx, 16, 32));
    const float m_new = fmaxf(m_run, mx);
    const float alpha = __expf(m_run - m_new);

    float p0[8], p1[8], ssum = 0.0f;
#pragma unroll
    for (int r = 0; r < 8; ++r) {
      p0[r] = __expf(t0[r] - m_new);
      p1[r] = __expf(t1[r] - m_new);
      ssum += p0[r] + p1[r];
    }
    ssum += __shfl_xor(ssum, 16, 32);
    l_run = l_run * alpha + ssum;
    m_run = m_new;

    // Pack P' C-tiles directly into the A fragment (pure per-lane pack).
    union { v16bf v; bf16 e[16]; } ap;
#pragma unroll
    for (int r = 0; r < 8; ++r) {
      ap.e[r]     = (bf16)p0[r];   // K offsets {0..7} / {8..15}
      ap.e[8 + r] = (bf16)p1[r];   // K offsets {16..23} / {24..31}
    }

    // Rescale accumulators: acc elem r belongs to query r (+8 in hi half).
    float af[8];
#pragma unroll
    for (int r = 0; r < 8; ++r) af[r] = __shfl(alpha, srcbase + r, 32);
#pragma unroll
    for (int t = 0; t < 4; ++t) {
#pragma unroll
      for (int r = 0; r < 8; ++r) acc[t][r] *= af[r];
    }

    // ctx += P'^T * V for the 4 d-tiles (V fragments from transposed V).
#pragma unroll
    for (int t = 0; t < 4; ++t) {
      v16bf bv = load_frag(Vt + (size_t)(t * 16) * S_ + n0, S_);
      acc[t] = wmma_bf16(ap.v, bv, acc[t]);
    }
  }

  // Normalize and store ctx into [B, S, NH*HD] bf16 for the output GEMM.
  float lf[8];
#pragma unroll
  for (int r = 0; r < 8; ++r) lf[r] = __shfl(l_run, srcbase + r, 32);
  const int rowadd = (lane & 16) ? 8 : 0;
  const int dl = lane & 15;
  bf16* cbase = ctx + ((size_t)bb * S_ + (q0 + rowadd)) * HID_ + h * HD_ + dl;
#pragma unroll
  for (int t = 0; t < 4; ++t) {
#pragma unroll
    for (int r = 0; r < 8; ++r)
      cbase[(size_t)r * HID_ + t * 16] = (bf16)(acc[t][r] / lf[r]);
  }
}

// ---------------------------------------------------------------------------
// Launch
// ---------------------------------------------------------------------------
extern "C" void kernel_launch(void* const* d_in, const int* in_sizes, int n_in,
                              void* d_out, int out_size, void* d_ws, size_t ws_size,
                              hipStream_t stream) {
  (void)in_sizes; (void)n_in; (void)out_size; (void)ws_size;
  const float* q_in = (const float*)d_in[0];
  const float* k_in = (const float*)d_in[1];
  const float* v_in = (const float*)d_in[2];
  const float* Wq = (const float*)d_in[3];
  const float* bq = (const float*)d_in[4];
  const float* Wk = (const float*)d_in[5];
  const float* bk = (const float*)d_in[6];
  const float* Wv = (const float*)d_in[7];
  const float* bv = (const float*)d_in[8];
  const float* Wo = (const float*)d_in[9];
  const float* bo = (const float*)d_in[10];

  char* ws = (char*)d_ws;
  const size_t SZ_X = (size_t)M_ * K_ * 2;          // 8 MB bf16 activations
  const size_t SZ_W = (size_t)K_ * HID_ * 2;        // 2 MB bf16 weight
  const size_t SZ_H = (size_t)B_ * NH_ * S_ * HD_ * 2;  // 8 MB per Q/K/Vt
  bf16* Xq  = (bf16*)(ws);
  bf16* Xk  = (bf16*)(ws + SZ_X);
  bf16* Xv  = (bf16*)(ws + 2 * SZ_X);
  bf16* Wtq = (bf16*)(ws + 3 * SZ_X);
  bf16* Wtk = (bf16*)(ws + 3 * SZ_X + SZ_W);
  bf16* Wtv = (bf16*)(ws + 3 * SZ_X + 2 * SZ_W);
  bf16* Wto = (bf16*)(ws + 3 * SZ_X + 3 * SZ_W);
  bf16* Qb  = (bf16*)(ws + 3 * SZ_X + 4 * SZ_W);
  bf16* Kb  = (bf16*)(ws + 3 * SZ_X + 4 * SZ_W + SZ_H);
  bf16* Vtb = (bf16*)(ws + 3 * SZ_X + 4 * SZ_W + 2 * SZ_H);
  bf16* Ctx = (bf16*)(ws + 3 * SZ_X + 4 * SZ_W + 3 * SZ_H);  // total 64 MB

  // 1) activations f32 -> bf16
  const int NX = M_ * K_;
  dim3 cg(NX / (256 * 4));
  cvt_f32_bf16<<<cg, 256, 0, stream>>>(q_in, Xq, NX);
  cvt_f32_bf16<<<cg, 256, 0, stream>>>(k_in, Xk, NX);
  cvt_f32_bf16<<<cg, 256, 0, stream>>>(v_in, Xv, NX);

  // 2) weights f32 [K x N] -> bf16 transposed [N x K]
  dim3 tg(HID_ / 32, K_ / 32), tb(32, 32);
  transpose_cvt<<<tg, tb, 0, stream>>>(Wq, Wtq);
  transpose_cvt<<<tg, tb, 0, stream>>>(Wk, Wtk);
  transpose_cvt<<<tg, tb, 0, stream>>>(Wv, Wtv);
  transpose_cvt<<<tg, tb, 0, stream>>>(Wo, Wto);

  // 3) QKV projections (fused bias + head-split / V-transpose epilogues)
  dim3 gg(M_ / 128, HID_ / 64);
  gemm_wmma<0><<<gg, 128, 0, stream>>>(Xq, Wtq, bq, Qb);
  gemm_wmma<0><<<gg, 128, 0, stream>>>(Xk, Wtk, bk, Kb);
  gemm_wmma<1><<<gg, 128, 0, stream>>>(Xv, Wtv, bv, Vtb);

  // 4) attention: one wave per 16 query rows per head
  dim3 ag(S_ / 16, B_ * NH_);
  attn_wmma<<<ag, 32, 0, stream>>>(Qb, Kb, Vtb, Ctx);

  // 5) output projection -> f32 d_out
  gemm_wmma<2><<<gg, 128, 0, stream>>>(Ctx, Wto, bo, d_out);
}
